// MultiHeadOPTAttention_88536455840276
// MI455X (gfx1250) — matrix operand order repaired and checked by reading, not compiled, not run on hardware
//
#include <hip/hip_runtime.h>
#include <hip/hip_bf16.h>

#define B_ 8
#define T_ 1024
#define E_ 128
#define H_ 8
#define NEGV (-1.0e9f)

typedef __attribute__((ext_vector_type(16))) _Float16 v16h;
typedef __attribute__((ext_vector_type(8)))  _Float16 v8h;
typedef __attribute__((ext_vector_type(8)))  float    v8f;
typedef unsigned int u32x4 __attribute__((ext_vector_type(4)));
typedef int          i32x8 __attribute__((ext_vector_type(8)));
typedef int          i32x4 __attribute__((ext_vector_type(4)));

// TDM builtin arity differs between toolchains (clang-22: 5 args, clang-23: 6)
#if defined(__clang_major__) && (__clang_major__ >= 23)
#define TDM_LOAD(g0, g1, g2, g3) \
  __builtin_amdgcn_tensor_load_to_lds(g0, g1, g2, g3, (i32x8){0,0,0,0,0,0,0,0}, 0)
#else
#define TDM_LOAD(g0, g1, g2, g3) \
  __builtin_amdgcn_tensor_load_to_lds(g0, g1, g2, g3, 0)
#endif

static __device__ __forceinline__ v8f wmma_f16(v16h a, v16h b, v8f c) {
  return __builtin_amdgcn_wmma_f32_16x16x32_f16(false, a, false, b, (short)0, c,
                                                false, false);
}

static __device__ __forceinline__ v16h mk16(v8h lo, v8h hi) {
  v16h r;
#pragma unroll
  for (int i = 0; i < 8; ++i) { r[i] = lo[i]; r[i + 8] = hi[i]; }
  return r;
}

// ---------------------------------------------------------------------------
// Kernel 0: one-time weight transpose+convert: wT[which][n][k] = w[k][n] (f16)
// ---------------------------------------------------------------------------
__global__ void wconv_kernel(const float* __restrict__ w_q,
                             const float* __restrict__ w_k,
                             const float* __restrict__ w_v,
                             _Float16* __restrict__ wT) {
  int idx   = blockIdx.x * 256 + threadIdx.x;  // 3 * 1024 * 128 total
  int which = idx >> 17;                       // 131072 elements per tensor
  int rem   = idx & 131071;
  int n     = rem >> 7;
  int k     = rem & 127;
  const float* w = (which == 0) ? w_q : (which == 1) ? w_k : w_v;
  wT[idx] = (_Float16)w[(size_t)k * (H_ * E_) + n];
}

// ---------------------------------------------------------------------------
// Kernel 1: QKV projections with WMMA.
//   Q,K : [h][b][t][e] f16 ;  Vt : [h][b][e][t] f16 (transposed for attn@V B)
// grid = (512 m-tiles, 8 heads, 3 {q,k,v}), 256 threads (8 waves).
// ---------------------------------------------------------------------------
__global__ void proj_qkv_kernel(const float* __restrict__ x,
                                const _Float16* __restrict__ wT,
                                _Float16* __restrict__ Qo,
                                _Float16* __restrict__ Ko,
                                _Float16* __restrict__ Vt) {
  __shared__ _Float16 lx[16][136];  // x tile f16 (row = 272B, 16B multiple)
  const int lane  = threadIdx.x & 31;
  const int wav   = threadIdx.x >> 5;
  const int m0    = blockIdx.x * 16;
  const int h     = blockIdx.y;
  const int which = blockIdx.z;

  for (int i = threadIdx.x; i < 16 * 128; i += 256) {
    int r = i >> 7, c = i & 127;
    lx[r][c] = (_Float16)x[(size_t)(m0 + r) * E_ + c];
  }
  __syncthreads();

  const int n0 = h * E_ + wav * 16;
  const _Float16* wp = wT + (size_t)which * (H_ * E_ * E_) +
                       (size_t)(n0 + (lane & 15)) * E_ + ((lane >> 4) << 4);
  v8f acc = {};
#pragma unroll
  for (int kk = 0; kk < 4; ++kk) {
    const int arow = lane & 15;
    const int kb   = kk * 32 + ((lane >> 4) << 4);
    v8h a0 = *(const v8h*)&lx[arow][kb];
    v8h a1 = *(const v8h*)&lx[arow][kb + 8];
    v8h b0 = *(const v8h*)(wp + kk * 32);
    v8h b1 = *(const v8h*)(wp + kk * 32 + 8);
    acc = wmma_f16(mk16(a0, a1), mk16(b0, b1), acc);
  }

  // ---- store epilogue: which-branch hoisted, shared address math ----
  const int mg0   = m0 + ((lane >> 4) << 3);   // all 8 v share the same b-index
  const int bidx  = mg0 >> 10;
  const int tidx0 = mg0 & (T_ - 1);
  const int e     = wav * 16 + (lane & 15);
  if (which == 2) {
    // Vt: the 8 accumulator elements are 8 consecutive t -> one b128 store
    v8h pk;
#pragma unroll
    for (int v = 0; v < 8; ++v) pk[v] = (_Float16)acc[v];
    *(v8h*)&Vt[(((size_t)h * B_ + bidx) * E_ + e) * T_ + tidx0] = pk;
  } else {
    _Float16* dst = ((which == 0) ? Qo : Ko) +
                    (((size_t)h * B_ + bidx) * T_ + tidx0) * E_ + e;
#pragma unroll
    for (int v = 0; v < 8; ++v) dst[(size_t)v * E_] = (_Float16)acc[v];
  }
}

// ---------------------------------------------------------------------------
// Kernel 2: sel = softmax(mean_t(x) @ w_s^T + b_s) -> [b][h]
// ---------------------------------------------------------------------------
__global__ void sel_kernel(const float* __restrict__ x,
                           const float* __restrict__ w_s,
                           const float* __restrict__ b_s,
                           float* __restrict__ sel) {
  __shared__ float mean[E_];
  __shared__ float logit[H_];
  const int b = blockIdx.x;
  const int e = threadIdx.x;
  float s = 0.f;
  for (int t = 0; t < T_; ++t) s += x[((size_t)b * T_ + t) * E_ + e];
  mean[e] = s * (1.0f / T_);
  __syncthreads();
  if (e < H_) {
    float lg = b_s[e];
    for (int j = 0; j < E_; ++j) lg += mean[j] * w_s[e * E_ + j];
    logit[e] = lg;
  }
  __syncthreads();
  if (e == 0) {
    float mx = logit[0];
    for (int hh = 1; hh < H_; ++hh) mx = fmaxf(mx, logit[hh]);
    float ssum = 0.f, ex[H_];
    for (int hh = 0; hh < H_; ++hh) { ex[hh] = expf(logit[hh] - mx); ssum += ex[hh]; }
    for (int hh = 0; hh < H_; ++hh) sel[b * H_ + hh] = ex[hh] / ssum;
  }
}

// ---------------------------------------------------------------------------
// Kernel 3: fused masked-sparsemax attention + gated head mix.
// grid = (64 q-tiles, 8 batches), 256 threads (8 waves).
// LDS: 64KB mask tile (TDM-staged once, reused by all 8 heads)
//    + 64.25KB f32 score strip + 32.5KB f16 attn strip.
// ---------------------------------------------------------------------------
__global__ void attn_kernel(const _Float16* __restrict__ Q,
                            const _Float16* __restrict__ K,
                            const _Float16* __restrict__ Vt,
                            const int* __restrict__ mask,
                            const float* __restrict__ sel,
                            float* __restrict__ out) {
  extern __shared__ char smem[];
  int*      lmask = (int*)smem;                                      // 65536 B
  float*    sc    = (float*)(smem + 16 * 1024 * 4);                  // 65792 B
  _Float16* at    = (_Float16*)(smem + 16 * 1024 * 4 + 16 * 1028 * 4);  // 33280 B
  const int SROW = 1028;
  const int AROW = 1040;
  __shared__ float s_tau[16];

  const int lane = threadIdx.x & 31;
  const int wav  = threadIdx.x >> 5;
  const int b    = blockIdx.y;
  const int q0   = blockIdx.x * 16;
  const float scale = 0.088388347648318447f;  // 1/sqrt(128)

  // ---- TDM: DMA mask[b][q0..q0+15][0..1023] (16x1024 i32) into LDS ----
  if (wav == 0) {
    unsigned long long ga =
        (unsigned long long)(const void*)(mask + ((size_t)b * T_ + q0) * T_);
    unsigned lds_off = (unsigned)(unsigned long long)(void*)lmask;
    u32x4 g0;
    i32x8 g1;
    i32x4 gz = {0, 0, 0, 0};
    g0[0] = 1u;                                  // count=1, user descriptor
    g0[1] = lds_off;                             // lds_addr (bytes)
    g0[2] = (unsigned)(ga & 0xFFFFFFFFu);        // global_addr[31:0]
    g0[3] = (unsigned)((ga >> 32) & 0x01FFFFFFu) | (2u << 30);  // hi | type=2
    g1[0] = (2 << 16);                           // data_size = 4 bytes
    g1[1] = (int)((T_ & 0xFFFF) << 16);          // tensor_dim0 = 1024 (lo16)
    g1[2] = (int)((T_ >> 16) | (16 << 16));      // dim0 hi | tensor_dim1 = 16
    g1[3] = (int)(T_ << 16);                     // tile_dim0 = 1024
    g1[4] = 16;                                  // tile_dim1 = 16
    g1[5] = T_;                                  // tensor_dim0_stride = 1024
    g1[6] = 0;
    g1[7] = 0;
    TDM_LOAD(g0, g1, gz, gz);
    __builtin_amdgcn_s_wait_tensorcnt(0);
  }
  __syncthreads();

  v8f facc = {};

  for (int h = 0; h < H_; ++h) {
    const float g   = sel[b * H_ + h];
    const size_t hb = (size_t)h * B_ + b;

    // ---- Q fragments kept in VGPRs across all 64 n-tiles ----
    v16h qf[4];
    {
      const _Float16* qbase = Q + (hb * T_ + q0 + (lane & 15)) * E_;
#pragma unroll
      for (int kk = 0; kk < 4; ++kk) {
        int kb = kk * 32 + ((lane >> 4) << 4);
        qf[kk] = mk16(*(const v8h*)(qbase + kb), *(const v8h*)(qbase + kb + 8));
      }
    }

    // ---- scores = scale * Q K^T, mask==1 -> NEG, into LDS ----
    for (int nt = wav; nt < 64; nt += 8) {
      const int t0 = nt * 16;
      v8f c = {};
      const _Float16* kbase =
          K + (hb * T_ + t0 + (lane & 15)) * E_ + ((lane >> 4) << 4);
      __builtin_prefetch((const void*)(kbase + 8 * 16 * E_), 0, 1);
#pragma unroll
      for (int kk = 0; kk < 4; ++kk) {
        v8h b0 = *(const v8h*)(kbase + kk * 32);
        v8h b1 = *(const v8h*)(kbase + kk * 32 + 8);
        c = wmma_f16(qf[kk], mk16(b0, b1), c);
      }
      const int tc = t0 + (lane & 15);
#pragma unroll
      for (int v = 0; v < 8; ++v) {
        int m   = ((lane >> 4) << 3) + v;
        int msk = lmask[m * T_ + tc];
        sc[m * SROW + tc] = msk ? NEGV : c[v] * scale;
      }
    }
    __syncthreads();

    // ---- sparsemax threshold tau per row via bisection (2 rows / wave) ----
    for (int r = wav * 2; r < wav * 2 + 2; ++r) {
      float z[32];
#pragma unroll
      for (int j = 0; j < 32; ++j) z[j] = sc[r * SROW + lane + 32 * j];
      float mx = z[0];
#pragma unroll
      for (int j = 1; j < 32; ++j) mx = fmaxf(mx, z[j]);
#pragma unroll
      for (int d = 16; d >= 1; d >>= 1) mx = fmaxf(mx, __shfl_xor(mx, d, 32));
      float tau;
      if (mx < -1.0e8f) {
        tau = mx + 1.0f;  // fully masked row -> zeros (matches reference)
      } else {
        float lo = mx - 1.0f, hi = mx;  // sum(max(z-tau,0))=1 bracketed here
        for (int it = 0; it < 30; ++it) {
          float mid = 0.5f * (lo + hi);
          float s = 0.f;
#pragma unroll
          for (int j = 0; j < 32; ++j) s += fmaxf(z[j] - mid, 0.f);
#pragma unroll
          for (int d = 16; d >= 1; d >>= 1) s += __shfl_xor(s, d, 32);
          if (s > 1.0f) lo = mid; else hi = mid;
        }
        tau = 0.5f * (lo + hi);
      }
      if (lane == 0) s_tau[r] = tau;
    }
    __syncthreads();

    // ---- single packed-convert pass: attn16 = max(sc - tau, 0) as f16 ----
    for (int i = threadIdx.x * 4; i < 16 * 1024; i += 256 * 4) {
      int r = i >> 10, c = i & 1023;
      float t = s_tau[r];
#pragma unroll
      for (int j = 0; j < 4; ++j)
        at[r * AROW + c + j] = (_Float16)fmaxf(sc[r * SROW + c + j] - t, 0.f);
    }
    __syncthreads();

    // ---- out_tile = attn @ V via WMMA (A from f16 LDS, B contiguous) ----
    v8f c2 = {};
    const _Float16* arow = at + (lane & 15) * AROW;
    const _Float16* vb =
        Vt + (hb * E_ + wav * 16 + (lane & 15)) * T_ + ((lane >> 4) << 4);
    for (int kt = 0; kt < 32; ++kt) {
      const int kb = kt * 32 + ((lane >> 4) << 4);
      v8h a0 = *(const v8h*)(arow + kb);
      v8h a1 = *(const v8h*)(arow + kb + 8);
      v8h b0 = *(const v8h*)(vb + kt * 32);
      v8h b1 = *(const v8h*)(vb + kt * 32 + 8);
      c2 = wmma_f16(mk16(a0, a1), mk16(b0, b1), c2);
    }
#pragma unroll
    for (int v = 0; v < 8; ++v) facc[v] += g * c2[v];
    __syncthreads();  // sc/at reused next head
  }

  // ---- out[b][q0+m][e] ----
#pragma unroll
  for (int v = 0; v < 8; ++v) {
    int m = ((lane >> 4) << 3) + v;
    int e = wav * 16 + (lane & 15);
    out[((size_t)b * T_ + q0 + m) * E_ + e] = facc[v];
  }
}

// ---------------------------------------------------------------------------
extern "C" void kernel_launch(void* const* d_in, const int* in_sizes, int n_in,
                              void* d_out, int out_size, void* d_ws,
                              size_t ws_size, hipStream_t stream) {
  const float* x    = (const float*)d_in[0];
  const int*   mask = (const int*)d_in[1];
  const float* w_q  = (const float*)d_in[2];
  const float* w_k  = (const float*)d_in[3];
  const float* w_v  = (const float*)d_in[4];
  const float* w_s  = (const float*)d_in[5];
  const float* b_s  = (const float*)d_in[6];
  float* out = (float*)d_out;

  const size_t HBTE = (size_t)H_ * B_ * T_ * E_;  // 8M halves
  _Float16* Q  = (_Float16*)d_ws;
  _Float16* Kp = Q + HBTE;
  _Float16* Vt = Kp + HBTE;
  _Float16* wT = Vt + HBTE;                        // 3*1024*128 halves
  float* sel   = (float*)(wT + (size_t)3 * H_ * E_ * E_);

  wconv_kernel<<<dim3(1536), 256, 0, stream>>>(w_q, w_k, w_v, wT);
  proj_qkv_kernel<<<dim3((B_ * T_) / 16, H_, 3), 256, 0, stream>>>(
      x, wT, Q, Kp, Vt);
  sel_kernel<<<dim3(B_), 128, 0, stream>>>(x, w_s, b_s, sel);
  const size_t lds_bytes = 16 * 1024 * 4 + 16 * 1028 * 4 + 16 * 1040 * 2;
  attn_kernel<<<dim3(T_ / 16, B_), 256, lds_bytes, stream>>>(
      Q, Kp, Vt, mask, sel, out);
}
